// GreedyViG_36258113913052
// MI455X (gfx1250) — matrix operand order repaired and checked by reading, not compiled, not run on hardware
//
#include <hip/hip_runtime.h>
#include <math.h>

// ---------------------------------------------------------------------------
// Types for CDNA5 WMMA (wave32)
// ---------------------------------------------------------------------------
typedef __attribute__((ext_vector_type(16))) __bf16 v16bf;
typedef __attribute__((ext_vector_type(8)))  float  v8f;

union BF16x16 {
  v16bf v;
  unsigned short u[16];
  unsigned long long ull[4];
};

__device__ inline unsigned short f2bf(float f) {
  unsigned int u = __float_as_uint(f);
  u += 0x7FFFu + ((u >> 16) & 1u);   // round-to-nearest-even truncate to bf16
  return (unsigned short)(u >> 16);
}

__device__ inline float gelu_f(float v) {   // exact erf GELU (torch nn.GELU default)
  return 0.5f * v * (1.0f + erff(v * 0.70710678118654752440f));
}

// ---------------------------------------------------------------------------
// Pointwise conv (1x1) as GEMM:  Y[b,m,hw] = sum_k W[m,k] * X[b,k,hw]
// bf16 WMMA 16x16x32, f32 accumulate. Fused: bias, folded BN, GELU, residual.
// Block = 256 threads = 8 waves; block tile M=64 x N=128; wave tile 32x32.
// LDS tiles are stored fragment-major so each lane's 16 bf16 fragment
// elements are one contiguous 32-byte LDS read (wide ds_load instead of
// 16-bit gathers + v_mov_b16 packing).
// ---------------------------------------------------------------------------
#define GTM 64
#define GTN 128
#define GTK 32
#define BSTR 40   // Bs row stride in halves (80B: keeps 16B alignment, spreads banks)

// K permutation swapping middle two 8-blocks (unused: As keeps natural K order).
// lanes 0-15 hold k=0..15 at positions 0..15; lanes 16-31 hold k=16..31 (same in As and Bs).
__device__ inline int aperm(int k) {
  int bk = k >> 3;                               // 0,1,2,3
  int nb = ((bk & 1) << 1) | ((bk >> 1) & 1);    // 0,2,1,3  (swap blocks 1<->2)
  return nb * 8 + (k & 7);
}

__global__ __launch_bounds__(256)
void gemm_pw_k(const float* __restrict__ X, const float* __restrict__ Wt,
               const float* __restrict__ bias, const float* __restrict__ bns,
               const float* __restrict__ bnt, const float* __restrict__ res,
               float* __restrict__ Y, int M, int K, int HW, long long Ntot,
               int gelu_flag)
{
  __shared__ __align__(32) unsigned short As[GTM][GTK];    // 64x32, natural K, 64B rows
  __shared__ __align__(32) unsigned short Bs[GTN][BSTR];   // col-major (n,k), 80B rows

  const int tid  = threadIdx.x;
  const int lane = tid & 31;
  const int wid  = tid >> 5;
  const int wm   = wid >> 2;       // 0..1 -> 32-row slab
  const int wn   = wid & 3;        // 0..3 -> 32-col slab
  const int lrow = lane & 15;
  const int lhi  = lane >> 4;      // lane group 0/1

  const long long n0 = (long long)blockIdx.x * GTN;
  const int m0 = blockIdx.y * GTM;

  v8f zero = {0.f,0.f,0.f,0.f,0.f,0.f,0.f,0.f};
  v8f acc[2][2];
  acc[0][0]=zero; acc[0][1]=zero; acc[1][0]=zero; acc[1][1]=zero;

  for (int kk = 0; kk < K; kk += GTK) {
    // ---- stage A tile (weights, 64 x 32) fp32 -> bf16, natural K order ----
    // 2048 elems, 4 per thread per pass, float4 global loads (K mult of 32).
    for (int e = tid * 4; e < GTM * GTK; e += 1024) {
      int r = e >> 5, k = e & 31;          // k % 4 == 0
      int m = m0 + r;
      float4 v4 = make_float4(0.f, 0.f, 0.f, 0.f);
      if (m < M) v4 = *(const float4*)&Wt[(long long)m * K + kk + k];
      int pos = k;                         // 4 elems stay contiguous inside 8-block
      As[r][pos + 0] = f2bf(v4.x);
      As[r][pos + 1] = f2bf(v4.y);
      As[r][pos + 2] = f2bf(v4.z);
      As[r][pos + 3] = f2bf(v4.w);
    }
    // ---- stage B tile (activations, 32 x 128) fp32 -> bf16, transposed ----
    // 4096 elems, 16 per thread, float4 along n (contiguous hw) when possible.
    for (int e = tid * 4; e < GTK * GTN; e += 1024) {
      int k = e >> 7, c = e & 127;         // c % 4 == 0
      long long n = n0 + c;
      float4 v4 = make_float4(0.f, 0.f, 0.f, 0.f);
      if (n + 3 < Ntot) {
        long long b = n / HW;
        int hw = (int)(n - b * HW);
        const float* xp = &X[((long long)b * K + (kk + k)) * (long long)HW + hw];
        if (hw + 3 < HW) {
          v4 = *(const float4*)xp;
        } else {
          float t[4];
          for (int q = 0; q < 4; ++q) {
            long long nn = n + q;
            long long bb = nn / HW;
            int hh = (int)(nn - bb * HW);
            t[q] = X[((long long)bb * K + (kk + k)) * (long long)HW + hh];
          }
          v4 = make_float4(t[0], t[1], t[2], t[3]);
        }
      } else {
        float t[4] = {0.f, 0.f, 0.f, 0.f};
        for (int q = 0; q < 4; ++q) {
          long long nn = n + q;
          if (nn < Ntot) {
            long long bb = nn / HW;
            int hh = (int)(nn - bb * HW);
            t[q] = X[((long long)bb * K + (kk + k)) * (long long)HW + hh];
          }
        }
        v4 = make_float4(t[0], t[1], t[2], t[3]);
      }
      Bs[c + 0][k] = f2bf(v4.x);
      Bs[c + 1][k] = f2bf(v4.y);
      Bs[c + 2][k] = f2bf(v4.z);
      Bs[c + 3][k] = f2bf(v4.w);
    }
    __syncthreads();

    // ---- fragments: one contiguous 32B LDS read per fragment ----
    BF16x16 afr[2], bfr[2];
#pragma unroll
    for (int i = 0; i < 2; ++i) {
      const unsigned long long* src =
          (const unsigned long long*)&As[wm * 32 + i * 16 + lrow][lhi * 16];
      afr[i].ull[0] = src[0]; afr[i].ull[1] = src[1];
      afr[i].ull[2] = src[2]; afr[i].ull[3] = src[3];
    }
#pragma unroll
    for (int j = 0; j < 2; ++j) {
      const unsigned long long* src =
          (const unsigned long long*)&Bs[wn * 32 + j * 16 + lrow][lhi * 16];
      bfr[j].ull[0] = src[0]; bfr[j].ull[1] = src[1];
      bfr[j].ull[2] = src[2]; bfr[j].ull[3] = src[3];
    }
#pragma unroll
    for (int i = 0; i < 2; ++i)
#pragma unroll
      for (int j = 0; j < 2; ++j)
        acc[i][j] = __builtin_amdgcn_wmma_f32_16x16x32_bf16(
            false, afr[i].v, false, bfr[j].v, (short)0, acc[i][j], false, false);
    __syncthreads();
  }

  // ---- epilogue: C/D layout -> m = base + r + 8*lhi, n = base + (lane&15) ----
#pragma unroll
  for (int i = 0; i < 2; ++i)
#pragma unroll
    for (int j = 0; j < 2; ++j)
#pragma unroll
      for (int r = 0; r < 8; ++r) {
        int m = m0 + wm * 32 + i * 16 + r + lhi * 8;
        long long n = n0 + wn * 32 + j * 16 + lrow;
        if (m < M && n < Ntot) {
          float v = acc[i][j][r] + bias[m];
          if (bns) v = v * bns[m] + bnt[m];
          if (gelu_flag) v = gelu_f(v);
          long long b = n / HW;
          int hw = (int)(n - b * HW);
          long long oi = ((long long)b * M + m) * (long long)HW + hw;
          if (res) v += res[oi];
          Y[oi] = v;
        }
      }
}

// ---------------------------------------------------------------------------
// Direct 3x3 conv, pad=1, stride 1/2, fused bias+BN(+GELU). (stem / downsample)
// ---------------------------------------------------------------------------
__global__ void conv3x3_k(const float* __restrict__ X, const float* __restrict__ Wt,
                          const float* __restrict__ bias, const float* __restrict__ bns,
                          const float* __restrict__ bnt, float* __restrict__ Y,
                          int Bn, int Cin, int Hin, int Win,
                          int Cout, int Hout, int Wout, int stride, int gelu_flag)
{
  long long idx = (long long)blockIdx.x * blockDim.x + threadIdx.x;
  long long tot = (long long)Bn * Cout * Hout * Wout;
  if (idx >= tot) return;
  int wo = (int)(idx % Wout); long long t = idx / Wout;
  int ho = (int)(t % Hout);   t /= Hout;
  int co = (int)(t % Cout);   int b = (int)(t / Cout);
  float acc = bias[co];
  int hi0 = ho * stride - 1, wi0 = wo * stride - 1;
  for (int ci = 0; ci < Cin; ++ci) {
    const float* xp = X + ((long long)b * Cin + ci) * Hin * Win;
    const float* wp = Wt + ((long long)co * Cin + ci) * 9;
#pragma unroll
    for (int kh = 0; kh < 3; ++kh) {
      int h = hi0 + kh;
      if (h < 0 || h >= Hin) continue;
#pragma unroll
      for (int kw = 0; kw < 3; ++kw) {
        int w = wi0 + kw;
        if (w < 0 || w >= Win) continue;
        acc += xp[(long long)h * Win + w] * wp[kh * 3 + kw];
      }
    }
  }
  if (bns) acc = acc * bns[co] + bnt[co];
  if (gelu_flag) acc = gelu_f(acc);
  Y[idx] = acc;
}

// ---------------------------------------------------------------------------
// Depthwise 3x3, pad=1, stride 1: bias + optional BN + optional GELU + residual.
// Used by IR depthwise stage (BN+GELU) and PEG (residual add x).
// ---------------------------------------------------------------------------
__global__ void dwconv3x3_k(const float* __restrict__ X, const float* __restrict__ Wt,
                            const float* __restrict__ bias, const float* __restrict__ bns,
                            const float* __restrict__ bnt, const float* __restrict__ res,
                            float* __restrict__ Y, int Bn, int C, int H, int Wd,
                            int gelu_flag)
{
  long long idx = (long long)blockIdx.x * blockDim.x + threadIdx.x;
  long long tot = (long long)Bn * C * H * Wd;
  if (idx >= tot) return;
  int w = (int)(idx % Wd); long long t = idx / Wd;
  int h = (int)(t % H); t /= H;
  int c = (int)(t % C); int b = (int)(t / C);
  const float* xp = X + ((long long)b * C + c) * H * Wd;
  const float* wp = Wt + (long long)c * 9;
  float acc = bias[c];
#pragma unroll
  for (int kh = 0; kh < 3; ++kh) {
    int hh = h - 1 + kh;
    if (hh < 0 || hh >= H) continue;
#pragma unroll
    for (int kw = 0; kw < 3; ++kw) {
      int ww = w - 1 + kw;
      if (ww < 0 || ww >= Wd) continue;
      acc += xp[(long long)hh * Wd + ww] * wp[kh * 3 + kw];
    }
  }
  if (bns) acc = acc * bns[c] + bnt[c];
  if (gelu_flag) acc = gelu_f(acc);
  if (res) acc += res[idx];
  Y[idx] = acc;
}

// ---------------------------------------------------------------------------
// dymrconv helpers
// ---------------------------------------------------------------------------
// norm[b,hw] = sum_c |x - roll(x, H/2, W/2)|
__global__ void l1norm_k(const float* __restrict__ X, float* __restrict__ norm,
                         int Bn, int C, int H, int Wd)
{
  long long idx = (long long)blockIdx.x * blockDim.x + threadIdx.x;
  long long tot = (long long)Bn * H * Wd;
  if (idx >= tot) return;
  int hw = (int)(idx % ((long long)H * Wd));
  int b  = (int)(idx / ((long long)H * Wd));
  int h = hw / Wd, w = hw % Wd;
  int h2 = h - H / 2; if (h2 < 0) h2 += H;
  int w2 = w - Wd / 2; if (w2 < 0) w2 += Wd;
  const float* xp = X + (long long)b * C * H * Wd;
  float s = 0.f;
  for (int c = 0; c < C; ++c) {
    long long o = (long long)c * H * Wd;
    s += fabsf(xp[o + (long long)h * Wd + w] - xp[o + (long long)h2 * Wd + w2]);
  }
  norm[idx] = s;
}

// per-batch thr = mean - unbiased std   (one block per batch)
__global__ void stats_k(const float* __restrict__ norm, float* __restrict__ stats, int HW)
{
  __shared__ float s1[256], s2[256];
  int b = blockIdx.x, tid = threadIdx.x;
  float a = 0.f, q = 0.f;
  for (int i = tid; i < HW; i += 256) {
    float v = norm[(long long)b * HW + i];
    a += v; q += v * v;
  }
  s1[tid] = a; s2[tid] = q;
  __syncthreads();
  for (int st = 128; st > 0; st >>= 1) {
    if (tid < st) { s1[tid] += s1[tid + st]; s2[tid] += s2[tid + st]; }
    __syncthreads();
  }
  if (tid == 0) {
    float mean = s1[0] / HW;
    float var  = (s2[0] - (float)HW * mean * mean) / (float)(HW - 1);
    stats[b] = mean - sqrtf(fmaxf(var, 0.f));
  }
}

// per-pixel shift masks: 7 distinct H-rolls (each counted twice in mask_sum)
__global__ void masks_k(const float* __restrict__ X, const float* __restrict__ stats,
                        int* __restrict__ maskb, float* __restrict__ msum,
                        int Bn, int C, int H, int Wd, int K)
{
  long long idx = (long long)blockIdx.x * blockDim.x + threadIdx.x;
  long long tot = (long long)Bn * H * Wd;
  if (idx >= tot) return;
  int hw = (int)(idx % ((long long)H * Wd));
  int b  = (int)(idx / ((long long)H * Wd));
  int h = hw / Wd, w = hw % Wd;
  const float* xp = X + (long long)b * C * H * Wd;
  float thr = stats[b];
  int NS = H / K;
  int bits = 0;
  for (int j = 0; j < NS; ++j) {
    int h2 = h - j * K; if (h2 < 0) h2 += H;
    float d = 0.f;
    for (int c = 0; c < C; ++c) {
      long long o = (long long)c * H * Wd;
      d += fabsf(xp[o + (long long)h * Wd + w] - xp[o + (long long)h2 * Wd + w]);
    }
    if (d < thr) bits |= (1 << j);
  }
  maskb[idx] = bits;
  msum[idx] = 2.0f * (float)__popc((unsigned)bits);   // shift list duplicated (H==W)
}

// per-batch wsel threshold = (max - min) * 0.4 of mask_sum
__global__ void minmax_k(const float* __restrict__ msum, float* __restrict__ stats, int HW)
{
  __shared__ float smn[256], smx[256];
  int b = blockIdx.x, tid = threadIdx.x;
  float mn = 3.4e38f, mx = -3.4e38f;
  for (int i = tid; i < HW; i += 256) {
    float v = msum[(long long)b * HW + i];
    mn = fminf(mn, v); mx = fmaxf(mx, v);
  }
  smn[tid] = mn; smx[tid] = mx;
  __syncthreads();
  for (int st = 128; st > 0; st >>= 1) {
    if (tid < st) { smn[tid] = fminf(smn[tid], smn[tid + st]);
                    smx[tid] = fmaxf(smx[tid], smx[tid + st]); }
    __syncthreads();
  }
  if (tid == 0) stats[32 + b] = (smx[0] - smn[0]) * 0.4f;
}

// build concat [x, x_j, x*wsel] -> (B, 3C, HW)
__global__ void buildy_k(const float* __restrict__ X, const int* __restrict__ maskb,
                         const float* __restrict__ msum, const float* __restrict__ stats,
                         float* __restrict__ Yc, int Bn, int C, int H, int Wd, int K)
{
  long long idx = (long long)blockIdx.x * blockDim.x + threadIdx.x;
  long long HW = (long long)H * Wd;
  long long tot = (long long)Bn * C * HW;
  if (idx >= tot) return;
  int hw = (int)(idx % HW); long long t = idx / HW;
  int c = (int)(t % C); int b = (int)(t / C);
  int h = hw / Wd, w = hw % Wd;
  const float* xp = X + ((long long)b * C + c) * HW;
  float xv = xp[(long long)h * Wd + w];
  int bits = maskb[(long long)b * HW + hw];
  int NS = H / K;
  float xj = 0.f;
  for (int j = 0; j < NS; ++j) {
    if ((bits >> j) & 1) {
      int h2 = h - j * K; if (h2 < 0) h2 += H;
      float d = xp[(long long)h2 * Wd + w] - xv;
      xj = fmaxf(xj, d);
    }
  }
  float wsel = (msum[(long long)b * HW + hw] < stats[32 + b]) ? 1.f : 0.f;
  long long base = (long long)b * 3 * C * HW;
  Yc[base + (long long)c * HW + hw]           = xv;
  Yc[base + (long long)(C + c) * HW + hw]     = xj;
  Yc[base + (long long)(2 * C + c) * HW + hw] = xv * wsel;
}

// global average pool (B,C,HW) -> (B,C)
__global__ void avgpool_k(const float* __restrict__ X, float* __restrict__ Y,
                          int Bn, int C, int HW)
{
  int idx = blockIdx.x * blockDim.x + threadIdx.x;
  if (idx >= Bn * C) return;
  const float* xp = X + (long long)idx * HW;
  float s = 0.f;
  for (int i = 0; i < HW; ++i) s += xp[i];
  Y[idx] = s / (float)HW;
}

// ---------------------------------------------------------------------------
// Host-side orchestration
// ---------------------------------------------------------------------------
struct ConvP { const float* b; const float* w; };
struct BnP   { const float* s; const float* t; };

struct StageP {
  ConvP f1, f2, fc1c, fc2c, mrc, peg, dsc;
  BnP fb1, fb2, fc1b, fc2b, mrb, dsb;
  ConvP ir_c1[6], ir_c2[6], ir_cd[6];
  BnP ir_b1[6], ir_b2[6], ir_bd[6];
  int nl;
};

extern "C" void kernel_launch(void* const* d_in, const int* in_sizes, int n_in,
                              void* d_out, int out_size, void* d_ws, size_t ws_size,
                              hipStream_t stream)
{
  (void)in_sizes; (void)n_in; (void)out_size; (void)ws_size;
  const int Bn = 32;
  const int CH[4] = {64, 128, 256, 512};
  const int NL[4] = {2, 2, 6, 2};
  const int KS[4] = {8, 4, 2, 1};

  // --- unpack params in jax tree_flatten order (sorted dict keys) ---
  int pidx = 0;
  auto F = [&]() -> const float* { return (const float*)d_in[pidx++]; };

  ConvP head; BnP predb; ConvP predc;
  head.b = F(); head.w = F();
  predb.s = F(); predb.t = F();
  predc.b = F(); predc.w = F();

  StageP st[4];
  for (int i = 0; i < 4; ++i) {
    StageP& s = st[i];
    // dagcs[0] (sorted keys)
    s.f1.b = F(); s.f1.w = F();
    s.f2.b = F(); s.f2.w = F();
    s.fb1.s = F(); s.fb1.t = F();
    s.fb2.s = F(); s.fb2.t = F();
    s.fc1b.s = F(); s.fc1b.t = F();
    s.fc1c.b = F(); s.fc1c.w = F();
    s.fc2b.s = F(); s.fc2b.t = F();
    s.fc2c.b = F(); s.fc2c.w = F();
    s.mrb.s = F(); s.mrb.t = F();
    s.mrc.b = F(); s.mrc.w = F();
    s.peg.b = F(); s.peg.w = F();
    if (i > 0) { s.dsb.s = F(); s.dsb.t = F(); s.dsc.b = F(); s.dsc.w = F(); }
    s.nl = NL[i];
    for (int r = 0; r < s.nl; ++r) {
      s.ir_b1[r].s = F(); s.ir_b1[r].t = F();
      s.ir_b2[r].s = F(); s.ir_b2[r].t = F();
      s.ir_bd[r].s = F(); s.ir_bd[r].t = F();
      s.ir_c1[r].b = F(); s.ir_c1[r].w = F();
      s.ir_c2[r].b = F(); s.ir_c2[r].w = F();
      s.ir_cd[r].b = F(); s.ir_cd[r].w = F();
    }
  }
  BnP sb1, sb2; ConvP sc1, sc2;
  sb1.s = F(); sb1.t = F();
  sb2.s = F(); sb2.t = F();
  sc1.b = F(); sc1.w = F();
  sc2.b = F(); sc2.w = F();
  const float* xin = F();

  // --- workspace bump allocator ---
  float* wsf = (float*)d_ws;
  size_t off = 0;
  auto alloc = [&](size_t n) { float* p = wsf + off; off += n; return p; };
  float* T1 = alloc(26000000ULL);
  float* T2 = alloc(26000000ULL);
  float* acts[3] = { alloc(6500000ULL), alloc(6500000ULL), alloc(6500000ULL) };
  float* NORM  = alloc(110000ULL);
  float* MSUM  = alloc(110000ULL);
  int*   MASKB = (int*)alloc(110000ULL);
  float* STATS = alloc(1024ULL);
  float* POOL  = alloc(20000ULL);
  float* PRED  = alloc(20000ULL);

  // --- launch helpers ---
  auto conv3 = [&](const float* X, ConvP c, const float* bs, const float* bt,
                   float* Y, int Cin, int Hin, int Win, int Cout, int stride, int gel) {
    int Hout = (stride == 2) ? Hin / 2 : Hin;
    int Wout = (stride == 2) ? Win / 2 : Win;
    long long tot = (long long)Bn * Cout * Hout * Wout;
    conv3x3_k<<<(unsigned)((tot + 255) / 256), 256, 0, stream>>>(
        X, c.w, c.b, bs, bt, Y, Bn, Cin, Hin, Win, Cout, Hout, Wout, stride, gel);
  };
  auto dwconv = [&](const float* X, ConvP c, const float* bs, const float* bt,
                    const float* res, float* Y, int C, int H, int gel) {
    long long tot = (long long)Bn * C * H * H;
    dwconv3x3_k<<<(unsigned)((tot + 255) / 256), 256, 0, stream>>>(
        X, c.w, c.b, bs, bt, res, Y, Bn, C, H, H, gel);
  };
  auto gemm = [&](const float* X, ConvP c, const BnP* bn, const float* res,
                  float* Y, int M, int K, int HW, int gel) {
    long long Ntot = (long long)Bn * HW;
    dim3 g((unsigned)((Ntot + GTN - 1) / GTN), (unsigned)((M + GTM - 1) / GTM));
    gemm_pw_k<<<g, 256, 0, stream>>>(X, c.w, c.b,
        bn ? bn->s : nullptr, bn ? bn->t : nullptr, res, Y, M, K, HW, Ntot, gel);
  };

  // --- stem ---
  conv3(xin, sc1, sb1.s, sb1.t, T1, 3, 224, 224, 32, 2, 1);       // (32,112,112)
  conv3(T1, sc2, sb2.s, sb2.t, acts[0], 32, 112, 112, 64, 2, 1);  // (64,56,56)
  int cur = 0;
  int H = 56;

  for (int i = 0; i < 4; ++i) {
    int C = CH[i];
    StageP& s = st[i];
    if (i > 0) {
      int d = (cur + 1) % 3;
      conv3(acts[cur], s.dsc, s.dsb.s, s.dsb.t, acts[d], CH[i - 1], H, H, C, 2, 0);
      cur = d; H /= 2;
    }
    int HW = H * H;

    // Inverted residuals
    for (int r = 0; r < s.nl; ++r) {
      gemm(acts[cur], s.ir_c1[r], &s.ir_b1[r], nullptr, T1, 4 * C, C, HW, 1);
      dwconv(T1, s.ir_cd[r], s.ir_bd[r].s, s.ir_bd[r].t, nullptr, T2, 4 * C, H, 1);
      int d = (cur + 1) % 3;
      gemm(T2, s.ir_c2[r], &s.ir_b2[r], acts[cur], acts[d], C, 4 * C, HW, 0);
      cur = d;
    }

    // DAGC (1 per stage)
    {
      int p = (cur + 1) % 3;
      // PEG: h = x + dwconv(x)
      dwconv(acts[cur], s.peg, nullptr, nullptr, acts[cur], acts[p], C, H, 0);
      // fc1
      gemm(acts[p], s.fc1c, &s.fc1b, nullptr, T1, C, C, HW, 0);
      // dymrconv on T1
      long long tot1 = (long long)Bn * HW;
      l1norm_k<<<(unsigned)((tot1 + 255) / 256), 256, 0, stream>>>(T1, NORM, Bn, C, H, H);
      stats_k<<<Bn, 256, 0, stream>>>(NORM, STATS, HW);
      masks_k<<<(unsigned)((tot1 + 255) / 256), 256, 0, stream>>>(
          T1, STATS, MASKB, MSUM, Bn, C, H, H, KS[i]);
      minmax_k<<<Bn, 256, 0, stream>>>(MSUM, STATS, HW);
      long long tot2 = (long long)Bn * C * HW;
      buildy_k<<<(unsigned)((tot2 + 255) / 256), 256, 0, stream>>>(
          T1, MASKB, MSUM, STATS, T2, Bn, C, H, H, KS[i]);
      gemm(T2, s.mrc, &s.mrb, nullptr, acts[p], C, 3 * C, HW, 1);
      int q = 3 - cur - p;
      gemm(acts[p], s.fc2c, &s.fc2b, acts[cur], acts[q], C, C, HW, 0);
      cur = q;
      // FFN
      gemm(acts[cur], s.f1, &s.fb1, nullptr, T1, 4 * C, C, HW, 1);
      int d2 = (cur + 1) % 3;
      gemm(T1, s.f2, &s.fb2, acts[cur], acts[d2], C, 4 * C, HW, 0);
      cur = d2;
    }
  }

  // --- pool + pred + head ---
  {
    int tot = Bn * 512;
    avgpool_k<<<(tot + 255) / 256, 256, 0, stream>>>(acts[cur], POOL, Bn, 512, H * H);
  }
  gemm(POOL, predc, &predb, nullptr, PRED, 512, 512, 1, 1);
  gemm(PRED, head, nullptr, nullptr, (float*)d_out, 1000, 512, 1, 0);
}
